// CrossModalAttention_83502754169155
// MI455X (gfx1250) — matrix operand order repaired and checked by reading, not compiled, not run on hardware
//
#include <hip/hip_runtime.h>

// ---------------------------------------------------------------------------
// CrossModalAttention for MI455X (gfx1250, wave32, WMMA 16x16x32 f16->f32)
//   B=4, N_s=4096, N_a=1024, D=256, H=8, hd=32
// ---------------------------------------------------------------------------
#define D_EMBED 256
#define HEADS   8
#define HD      32
#define NS      4096
#define NA      1024
#define BATCH   4
#define SM_SCALE 0.17677669529663687f   // 1/sqrt(32)

typedef __attribute__((ext_vector_type(16))) _Float16 v16h;
typedef __attribute__((ext_vector_type(8)))  float    v8f;

static __device__ __forceinline__ v8f wmma_f16f32(v16h a, v16h b, v8f c) {
  // D = A(16x32 f16) * B(32x16 f16) + C(16x16 f32)
  return __builtin_amdgcn_wmma_f32_16x16x32_f16(false, a, false, b,
                                                (short)0, c, false, false);
}

// ---- A-matrix fragment (16x32 f16): linear K ------------------------------
// lane L: row = r0 + L%16 ; halves 0..15 -> K = k0 + 16*(L/16) + i
static __device__ __forceinline__ v16h frag_a_f32(const float* src, int ld, int r0, int k0) {
  const int lane = threadIdx.x & 31;
  const float* p = src + (size_t)(r0 + (lane & 15)) * ld + k0 + ((lane >> 4) << 4);
  v16h f;
#pragma unroll
  for (int i = 0; i < 16; ++i) f[i] = (_Float16)p[i];
  return f;
}
static __device__ __forceinline__ v16h frag_a_h(const _Float16* src, int ld, int r0, int k0) {
  const int lane = threadIdx.x & 31;
  const _Float16* p = src + (size_t)(r0 + (lane & 15)) * ld + k0 + ((lane >> 4) << 4);
  v16h f;
#pragma unroll
  for (int i = 0; i < 16; ++i) f[i] = p[i];
  return f;
}
// ---- B-matrix fragment (32x16 f16): linear K ------------------------------
// lane L: col = n0 + L%16 ; halves 0..15 -> K = k0 + 16*(L/16) + i
// source is row-major [col][K] (i.e. B^T): one contiguous 32B chunk per lane.
static __device__ __forceinline__ v16h frag_b_h(const _Float16* src, int ld, int n0, int k0) {
  const int lane = threadIdx.x & 31;
  const _Float16* p = src + (size_t)(n0 + (lane & 15)) * ld + k0 + ((lane >> 4) << 4);
  v16h f;
#pragma unroll
  for (int i = 0; i < 16; ++i) f[i] = p[i];
  return f;
}

// ---------------------------------------------------------------------------
// Prologue: one-time fp32 -> f16 weight conversion (256x256 each).
// ---------------------------------------------------------------------------
__global__ __launch_bounds__(256) void cvt_w_f16(const float* __restrict__ src,
                                                 _Float16* __restrict__ dst) {
  const int i = (blockIdx.x * 256 + threadIdx.x) * 4;
#pragma unroll
  for (int j = 0; j < 4; ++j) dst[i + j] = (_Float16)src[i + j];
}

// ---------------------------------------------------------------------------
// Stage 1/3: GEMM  out[m,n] = sum_k A[m,k] * W[n,k] + bias[n]   (K=N=256)
// Per wave: one 16x64 strip (4 accumulators) -> A fragment reused 4x.
// mode 0: f16 natural [M,256]   mode 1: f16 head-transposed V [b,h,hd,NA]
// mode 2: f32 natural [M,256]
// 8 waves/block, grid = M/32 blocks (exact).
// ---------------------------------------------------------------------------
__global__ __launch_bounds__(256) void gemm256_wmma(
    const float* __restrict__ A, const _Float16* __restrict__ Wh,
    const float* __restrict__ bias, _Float16* __restrict__ outH,
    float* __restrict__ outF, int mode) {
  const int wid  = threadIdx.x >> 5;
  const int lane = threadIdx.x & 31;
  const int unit = blockIdx.x * 8 + wid;
  const int mt = unit >> 2;          // m-tile (16 rows)
  const int ng = unit & 3;           // n-group (64 cols)

  v8f acc[4] = {{}, {}, {}, {}};
#pragma unroll
  for (int k = 0; k < 256; k += 32) {
    const v16h af = frag_a_f32(A, 256, mt * 16, k);
#pragma unroll
    for (int j = 0; j < 4; ++j) {
      v16h bf = frag_b_h(Wh, 256, ng * 64 + j * 16, k);
      acc[j] = wmma_f16f32(af, bf, acc[j]);
    }
  }

  const int rb = mt * 16 + ((lane >> 4) << 3);
  if (mode == 0) {
#pragma unroll
    for (int j = 0; j < 4; ++j) {
      const int n = ng * 64 + j * 16 + (lane & 15);
      const float bv = bias[n];
#pragma unroll
      for (int r = 0; r < 8; ++r)
        outH[(size_t)(rb + r) * 256 + n] = (_Float16)(acc[j][r] + bv);
    }
  } else if (mode == 1) {
#pragma unroll
    for (int j = 0; j < 4; ++j) {
      const int n = ng * 64 + j * 16 + (lane & 15);
      const float bv = bias[n];
      const int h = n >> 5, v = n & 31;
#pragma unroll
      for (int r = 0; r < 8; ++r) {
        const int m = rb + r, b = m >> 10, a = m & 1023;   // M = B*NA
        outH[(size_t)((b * HEADS + h) * HD + v) * NA + a] = (_Float16)(acc[j][r] + bv);
      }
    }
  } else {
#pragma unroll
    for (int j = 0; j < 4; ++j) {
      const int n = ng * 64 + j * 16 + (lane & 15);
      const float bv = bias[n];
#pragma unroll
      for (int r = 0; r < 8; ++r)
        outF[(size_t)(rb + r) * 256 + n] = acc[j][r] + bv;
    }
  }
}

// ---------------------------------------------------------------------------
// Stage 2: per-wave 16-row attention strip for one (b,h).
// Pass 1: WMMA scores -> online row max / sumexp (16-lane shfl reductions).
// Pass 2: WMMA scores again -> p = exp(s-m)/l -> write attn map (fp32),
//         bounce p(f16) through LDS into A-layout, WMMA-accumulate P*V.
// 4 waves / block, grid = B*H*(NS/16)/4 = 2048 blocks (exact).
// ---------------------------------------------------------------------------
__global__ __launch_bounds__(128) void attn_wmma(
    const _Float16* __restrict__ Qh, const _Float16* __restrict__ Kh,
    const _Float16* __restrict__ Vt, float* __restrict__ attn,
    float* __restrict__ attended) {
  __shared__ _Float16 pstage[4][16 * 32];
  const int wid  = threadIdx.x >> 5;
  const int lane = threadIdx.x & 31;
  const int wt = blockIdx.x * 4 + wid;        // 0..8191
  const int st = wt & 255;                    // s-tile
  const int bh = wt >> 8;                     // b*8+h
  const int b = bh >> 3, h = bh & 7;

  const _Float16* Qb = Qh + (size_t)b * NS * 256;
  const _Float16* Kb = Kh + (size_t)b * NA * 256;
  const _Float16* Vb = Vt + (size_t)bh * HD * NA;

  const v16h qf = frag_a_h(Qb, 256, st * 16, h * HD);

  // ---- pass 1: softmax statistics (per-lane-half holds rows 8*(lane/16)+r)
  float m[8], l[8];
#pragma unroll
  for (int r = 0; r < 8; ++r) { m[r] = -1e30f; l[r] = 0.0f; }
  for (int ct = 0; ct < NA / 16; ++ct) {
    v16h kf = frag_b_h(Kb, 256, ct * 16, h * HD);
    v8f z = {};
    v8f s = wmma_f16f32(qf, kf, z);
#pragma unroll
    for (int r = 0; r < 8; ++r) {
      float x = s[r] * SM_SCALE;
      float mx = x;
#pragma unroll
      for (int o = 1; o < 16; o <<= 1) mx = fmaxf(mx, __shfl_xor(mx, o, 32));
      const float mn = fmaxf(m[r], mx);
      float e = __expf(x - mn);
#pragma unroll
      for (int o = 1; o < 16; o <<= 1) e += __shfl_xor(e, o, 32);
      l[r] = l[r] * __expf(m[r] - mn) + e;
      m[r] = mn;
    }
  }
  float il[8];
#pragma unroll
  for (int r = 0; r < 8; ++r) il[r] = 1.0f / l[r];

  // ---- pass 2: recompute, normalize, emit attn map, accumulate P*V
  v8f o0 = {}, o1 = {};
  _Float16* pst = pstage[wid];
  float* ab = attn + ((size_t)bh * NS + st * 16) * NA;
  const int cn = lane & 15, hh = lane >> 4;
  for (int cp = 0; cp < NA / 32; ++cp) {
#pragma unroll
    for (int t = 0; t < 2; ++t) {
      const int ct = cp * 2 + t;
      v16h kf = frag_b_h(Kb, 256, ct * 16, h * HD);
      v8f z = {};
      v8f s = wmma_f16f32(qf, kf, z);
#pragma unroll
      for (int r = 0; r < 8; ++r) {
        const float p = __expf(s[r] * SM_SCALE - m[r]) * il[r];
        const int row = r + 8 * hh;
        ab[(size_t)row * NA + ct * 16 + cn] = p;        // attn_weights output
        pst[row * 32 + t * 16 + cn] = (_Float16)p;      // LDS bounce (C->A layout)
      }
    }
    asm volatile("s_wait_dscnt 0" ::: "memory");        // LDS writes visible in-wave
    v16h pf = frag_a_h(pst, 32, 0, 0);
    v16h v0 = frag_b_h(Vb, NA, 0,  cp * 32);
    v16h v1 = frag_b_h(Vb, NA, 16, cp * 32);
    o0 = wmma_f16f32(pf, v0, o0);
    o1 = wmma_f16f32(pf, v1, o1);
  }

  float* att = attended + ((size_t)b * NS + st * 16) * 256 + h * HD;
#pragma unroll
  for (int r = 0; r < 8; ++r) {
    const int row = r + 8 * hh;
    att[(size_t)row * 256 + cn]      = o0[r];
    att[(size_t)row * 256 + 16 + cn] = o1[r];
  }
}

// ---------------------------------------------------------------------------
// Stage 4: residual + LayerNorm, one wave per row (256 cols = 8/lane).
// ---------------------------------------------------------------------------
__global__ __launch_bounds__(256) void resid_ln(
    const float* __restrict__ proj, const float* __restrict__ resid,
    const float* __restrict__ gamma, const float* __restrict__ beta,
    float* __restrict__ out) {
  const int wid = threadIdx.x >> 5, lane = threadIdx.x & 31;
  const size_t row = (size_t)blockIdx.x * 8 + wid;
  const float* pr = proj + row * 256 + lane * 8;
  const float* rr = resid + row * 256 + lane * 8;
  float x[8], s = 0.0f, s2 = 0.0f;
#pragma unroll
  for (int i = 0; i < 8; ++i) {
    x[i] = pr[i] + rr[i];
    s += x[i]; s2 += x[i] * x[i];
  }
#pragma unroll
  for (int o = 1; o < 32; o <<= 1) { s += __shfl_xor(s, o, 32); s2 += __shfl_xor(s2, o, 32); }
  const float mu = s * (1.0f / 256.0f);
  const float var = s2 * (1.0f / 256.0f) - mu * mu;
  const float rs = rsqrtf(var + 1e-5f);
  float* po = out + row * 256 + lane * 8;
#pragma unroll
  for (int i = 0; i < 8; ++i) {
    const int c = lane * 8 + i;
    po[i] = (x[i] - mu) * rs * gamma[c] + beta[c];
  }
}

// ---------------------------------------------------------------------------
extern "C" void kernel_launch(void* const* d_in, const int* in_sizes, int n_in,
                              void* d_out, int out_size, void* d_ws, size_t ws_size,
                              hipStream_t stream) {
  const float* sat = (const float*)d_in[0];   // (4,4096,256)
  const float* aux = (const float*)d_in[1];   // (4,1024,256)
  const float* Wq = (const float*)d_in[2];  const float* bq = (const float*)d_in[3];
  const float* Wk = (const float*)d_in[4];  const float* bk = (const float*)d_in[5];
  const float* Wv = (const float*)d_in[6];  const float* bv = (const float*)d_in[7];
  const float* Wo = (const float*)d_in[8];  const float* bo = (const float*)d_in[9];
  const float* ga = (const float*)d_in[10]; const float* be = (const float*)d_in[11];

  const size_t Mq = (size_t)BATCH * NS;   // 16384
  const size_t Ma = (size_t)BATCH * NA;   // 4096
  const size_t WE = (size_t)256 * 256;    // weight elements

  char* ws = (char*)d_ws;
  _Float16* Wqh = (_Float16*)ws;                    // 4 x 128 KB f16 weights
  _Float16* Wkh = Wqh + WE;
  _Float16* Wvh = Wkh + WE;
  _Float16* Woh = Wvh + WE;
  _Float16* Qh  = Woh + WE;                         // 8 MB
  _Float16* Kh  = Qh + Mq * 256;                    // 2 MB
  _Float16* Vt  = Kh + Ma * 256;                    // 2 MB
  float* attended = (float*)(Vt + Ma * 256);        // 16 MB
  float* proj     = attended + Mq * 256;            // 16 MB

  float* out  = (float*)d_out;                 // (4,4096,256)
  float* attn = out + Mq * 256;                // (4,8,4096,1024)

  // Stage 0: one-time weight conversion to f16
  cvt_w_f16<<<dim3(64), 256, 0, stream>>>(Wq, Wqh);
  cvt_w_f16<<<dim3(64), 256, 0, stream>>>(Wk, Wkh);
  cvt_w_f16<<<dim3(64), 256, 0, stream>>>(Wv, Wvh);
  cvt_w_f16<<<dim3(64), 256, 0, stream>>>(Wo, Woh);
  // Stage 1: projections (WMMA, 16x64 strip per wave)
  gemm256_wmma<<<dim3(Mq / 32), 256, 0, stream>>>(sat, Wqh, bq, Qh, nullptr, 0);
  gemm256_wmma<<<dim3(Ma / 32), 256, 0, stream>>>(aux, Wkh, bk, Kh, nullptr, 0);
  gemm256_wmma<<<dim3(Ma / 32), 256, 0, stream>>>(aux, Wvh, bv, Vt, nullptr, 1);
  // Stage 2: attention + attn-map emission (WMMA)
  attn_wmma<<<dim3(BATCH * HEADS * (NS / 16) / 4), 128, 0, stream>>>(Qh, Kh, Vt, attn, attended);
  // Stage 3: output projection (WMMA)
  gemm256_wmma<<<dim3(Mq / 32), 256, 0, stream>>>(attended, Woh, bo, nullptr, proj, 2);
  // Stage 4: residual + LayerNorm
  resid_ln<<<dim3(Mq / 8), 256, 0, stream>>>(proj, sat, ga, be, out);
}
